// QuantizedLinearLayerWithActivation_13091060318662
// MI455X (gfx1250) — 8-bit matrix operand order repaired and checked by reading, compiled, not run on hardware
//
#include <hip/hip_runtime.h>
#include <stdint.h>


typedef int v8i __attribute__((ext_vector_type(8)));
typedef int v4i_vs __attribute__((__vector_size__(16)));

constexpr int BM = 128;
constexpr int BN = 128;
constexpr int BK = 128;
constexpr int ROWB = BK + 16;

#define ACT_INV_SCALE (1.0f / 0.02f)

#if __has_builtin(__builtin_amdgcn_global_load_async_to_lds_b128)
#define HAVE_ASYNC_LDS 1
#else
#define HAVE_ASYNC_LDS 0
#endif

typedef __attribute__((address_space(1))) v4i_vs gv4i_t;
typedef __attribute__((address_space(3))) v4i_vs lv4i_t;

__device__ __forceinline__ int quant1(float v) {
    float r = rintf(v * ACT_INV_SCALE);
    r = fminf(fmaxf(r, -32768.0f), 32767.0f);
    return (int)r;
}

__global__ __launch_bounds__(256) void quant_x_kernel(
    const float* __restrict__ x, uint32_t* __restrict__ lo,
    uint32_t* __restrict__ hi, int n4) {
    int idx = blockIdx.x * blockDim.x + threadIdx.x;
    if (idx >= n4) return;
    float4 v = ((const float4*)x)[idx];
    int a = quant1(v.x), b = quant1(v.y), c = quant1(v.z), d = quant1(v.w);
    lo[idx] = (uint32_t)(a & 255) | ((uint32_t)(b & 255) << 8) |
              ((uint32_t)(c & 255) << 16) | ((uint32_t)(d & 255) << 24);
    hi[idx] = (uint32_t)((a >> 8) & 255) | ((uint32_t)((b >> 8) & 255) << 8) |
              ((uint32_t)((c >> 8) & 255) << 16) | ((uint32_t)((d >> 8) & 255) << 24);
}

__global__ __launch_bounds__(256) void quant_w_kernel(
    const int* __restrict__ w, uint32_t* __restrict__ w8, int n4) {
    int idx = blockIdx.x * blockDim.x + threadIdx.x;
    if (idx >= n4) return;
    int4 v = ((const int4*)w)[idx];
    w8[idx] = (uint32_t)(v.x & 255) | ((uint32_t)(v.y & 255) << 8) |
              ((uint32_t)(v.z & 255) << 16) | ((uint32_t)(v.w & 255) << 24);
}

__device__ __forceinline__ void cp16(void* l, const void* g) {
#if HAVE_ASYNC_LDS
    __builtin_amdgcn_global_load_async_to_lds_b128((gv4i_t*)g, (lv4i_t*)l, 0, 0);
#else
    *(uint4*)l = *(const uint4*)g;
#endif
}

__device__ __forceinline__ void wait_stage() {
#if HAVE_ASYNC_LDS
#if __has_builtin(__builtin_amdgcn_s_wait_asynccnt)
    __builtin_amdgcn_s_wait_asynccnt(0);
#else
    asm volatile("s_wait_asynccnt 0x0" ::: "memory");
#endif
#endif
}

__device__ __forceinline__ void stage_chunk(
    unsigned char* __restrict__ dAlo, unsigned char* __restrict__ dAhi,
    unsigned char* __restrict__ dW,
    const unsigned char* __restrict__ gAlo, const unsigned char* __restrict__ gAhi,
    const unsigned char* __restrict__ gW, int K, int t) {
    const int row = t >> 1;
    const int half = (t & 1) << 6;
    const size_t go = (size_t)row * K + half;
    const int lo = row * ROWB + half;
#pragma unroll
    for (int i = 0; i < 4; ++i) {
        cp16(dAlo + lo + 16 * i, gAlo + go + 16 * i);
        cp16(dAhi + lo + 16 * i, gAhi + go + 16 * i);
        cp16(dW   + lo + 16 * i, gW   + go + 16 * i);
    }
}

__device__ __forceinline__ v8i frag_a(const unsigned char* p) {
    uint2 d0 = *(const uint2*)(p);
    uint2 d1 = *(const uint2*)(p + 16);
    uint2 d2 = *(const uint2*)(p + 32);
    uint2 d3 = *(const uint2*)(p + 48);
    v8i f;
    f[0] = (int)d0.x; f[1] = (int)d0.y;
    f[2] = (int)d1.x; f[3] = (int)d1.y;
    f[4] = (int)d2.x; f[5] = (int)d2.y;
    f[6] = (int)d3.x; f[7] = (int)d3.y;
    return f;
}

__device__ __forceinline__ v8i frag_b(const unsigned char* p) {
    uint2 d0 = *(const uint2*)(p);
    uint2 d1 = *(const uint2*)(p + 16);
    uint2 d2 = *(const uint2*)(p + 32);
    uint2 d3 = *(const uint2*)(p + 48);
    v8i f;
    f[0] = (int)d0.x; f[1] = (int)d0.y; f[2] = (int)d1.x; f[3] = (int)d1.y;
    f[4] = (int)d2.x; f[5] = (int)d2.y; f[6] = (int)d3.x; f[7] = (int)d3.y;
    return f;
}

__global__ __launch_bounds__(256) void gemm_iu8_kernel(
    const unsigned char* __restrict__ xlo, const unsigned char* __restrict__ xhi,
    const unsigned char* __restrict__ w8, const float* __restrict__ scale,
    const float* __restrict__ bias, float* __restrict__ out,
    int M, int N, int K) {

    __shared__ alignas(16) unsigned char sAlo[2][BM * ROWB];
    __shared__ alignas(16) unsigned char sAhi[2][BM * ROWB];
    __shared__ alignas(16) unsigned char sW[2][BN * ROWB];

    const int n0 = blockIdx.x * BN;
    const int m0 = blockIdx.y * BM;
    const int t = threadIdx.x;
    const int lane = t & 31;
    const int wv = t >> 5;
    const int wm = wv & 3;
    const int wn = wv >> 2;
    const int lhi = lane >> 4;
    const int l15 = lane & 15;

    const unsigned char* gAlo = xlo + (size_t)m0 * K;
    const unsigned char* gAhi = xhi + (size_t)m0 * K;
    const unsigned char* gW   = w8  + (size_t)n0 * K;

    v8i acc_lo[2][4];
    v8i acc_hi[2][4];
#pragma unroll
    for (int mt = 0; mt < 2; ++mt)
#pragma unroll
        for (int nt = 0; nt < 4; ++nt) {
#pragma unroll
            for (int r = 0; r < 8; ++r) { acc_lo[mt][nt][r] = 0; acc_hi[mt][nt][r] = 0; }
        }

    stage_chunk(sAlo[0], sAhi[0], sW[0], gAlo, gAhi, gW, K, t);

    int buf = 0;
    for (int kc = 0; kc < K; kc += BK, buf ^= 1) {
        wait_stage();
        __syncthreads();

        if (kc + BK < K) {
            stage_chunk(sAlo[buf ^ 1], sAhi[buf ^ 1], sW[buf ^ 1],
                        gAlo + kc + BK, gAhi + kc + BK, gW + kc + BK, K, t);
        }

        const unsigned char* bAlo = sAlo[buf];
        const unsigned char* bAhi = sAhi[buf];
        const unsigned char* bW   = sW[buf];

#pragma unroll
        for (int ks = 0; ks < BK; ks += 64) {
            const int aoff = ks + (lhi << 3);
            const int boff = ks + (lhi << 3);

            v8i alo[2], ahi[2];
#pragma unroll
            for (int mt = 0; mt < 2; ++mt) {
                const int arow = (wm * 32 + mt * 16 + l15) * ROWB;
                alo[mt] = frag_a(&bAlo[arow + aoff]);
                ahi[mt] = frag_a(&bAhi[arow + aoff]);
            }
            v8i bf[4];
#pragma unroll
            for (int nt = 0; nt < 4; ++nt) {
                const int brow = (wn * 64 + nt * 16 + l15) * ROWB;
                bf[nt] = frag_b(&bW[brow + boff]);
            }
#pragma unroll
            for (int mt = 0; mt < 2; ++mt) {
#pragma unroll
                for (int nt = 0; nt < 4; ++nt) {
                    acc_lo[mt][nt] = __builtin_amdgcn_wmma_i32_16x16x64_iu8(
                        false, alo[mt], true, bf[nt], acc_lo[mt][nt], false, false);
                    acc_hi[mt][nt] = __builtin_amdgcn_wmma_i32_16x16x64_iu8(
                        true, ahi[mt], true, bf[nt], acc_hi[mt][nt], false, false);
                }
            }
        }
        __syncthreads();
    }

#pragma unroll
    for (int nt = 0; nt < 4; ++nt) {
        const int n = n0 + wn * 64 + nt * 16 + l15;
        const float s = 0.02f * scale[n];
        const float b = bias[n];
#pragma unroll
        for (int mt = 0; mt < 2; ++mt) {
            const int mbase = m0 + wm * 32 + mt * 16 + (lhi << 3);
#pragma unroll
            for (int r = 0; r < 8; ++r) {
                long long iv = ((long long)acc_hi[mt][nt][r] << 8) +
                               (long long)acc_lo[mt][nt][r];
                float v = (float)iv * s + b;
                __builtin_nontemporal_store(v, &out[(size_t)(mbase + r) * N + n]);
            }
        }
    }
}

extern "C" void kernel_launch(void* const* d_in, const int* in_sizes, int n_in,
                              void* d_out, int out_size, void* d_ws, size_t ws_size,
                              hipStream_t stream) {
    const float* x     = (const float*)d_in[0];
    const int*   w     = (const int*)d_in[1];
    const float* scale = (const float*)d_in[2];
    const float* bias  = (const float*)d_in[3];
    float* out = (float*)d_out;

    const int OUT = in_sizes[2];
    const int K   = in_sizes[1] / OUT;
    const int M   = in_sizes[0] / K;

    unsigned char* xlo = (unsigned char*)d_ws;
    unsigned char* xhi = xlo + (size_t)M * K;
    unsigned char* w8p = xhi + (size_t)M * K;

    const int nx4 = (M * K) / 4;
    const int nw4 = (OUT * K) / 4;

    quant_x_kernel<<<(nx4 + 255) / 256, 256, 0, stream>>>(
        x, (uint32_t*)xlo, (uint32_t*)xhi, nx4);
    quant_w_kernel<<<(nw4 + 255) / 256, 256, 0, stream>>>(
        w, (uint32_t*)w8p, nw4);

    dim3 grid(OUT / BN, M / BM);
    gemm_iu8_kernel<<<grid, 256, 0, stream>>>(
        xlo, xhi, w8p, scale, bias, out, M, OUT, K);
}
